// QM9_enc_91285234909852
// MI455X (gfx1250) — matrix operand order repaired and checked by reading, not compiled, not run on hardware
//
#include <hip/hip_runtime.h>
#include <hip/hip_bf16.h>

typedef __bf16 bf16_t;
typedef __attribute__((ext_vector_type(16))) __bf16 v16bf;
typedef __attribute__((ext_vector_type(8)))  float  v8f;
typedef unsigned int u32x4 __attribute__((ext_vector_type(4)));
typedef int          i32x8 __attribute__((ext_vector_type(8)));
typedef int          i32x4 __attribute__((ext_vector_type(4)));

#if defined(__has_builtin)
#if __has_builtin(__builtin_amdgcn_tensor_load_to_lds) && \
    __has_builtin(__builtin_amdgcn_s_wait_tensorcnt)
#define HAVE_TDM 1
#endif
#endif

// ---------------------------------------------------------------------------
// WMMA helpers (CDNA5 gfx1250, wave32). Layouts per CDNA5 ISA 7.12.2
// ---------------------------------------------------------------------------
__device__ __forceinline__ v8f wmma_bf16(v16bf a, v16bf b, v8f c) {
  // (neg_a, A, neg_b, B, c_mod, C, reuse_a, reuse_b)
  return __builtin_amdgcn_wmma_f32_16x16x32_bf16(false, a, false, b, (short)0, c,
                                                 false, false);
}

// A tile is 16x32 (M x K), bf16, row-major in memory with given row stride.
// lane<16 : M=lane,    elements 0..7 -> K=0..7,  8..15 -> K=16..23
// lane>=16: M=lane-16, elements 0..7 -> K=8..15, 8..15 -> K=24..31
__device__ __forceinline__ v16bf load_a_frag(const bf16_t* base, int stride, int lane) {
  int m   = lane & 15;
  int khi = (lane & 16) ? 8 : 0;
  const bf16_t* rp = base + (size_t)m * stride + khi;
  v16bf a;
#pragma unroll
  for (int j = 0; j < 8; ++j) a[j] = rp[j];
#pragma unroll
  for (int j = 0; j < 8; ++j) a[8 + j] = rp[16 + j];
  return a;
}

// B tile is 32x16 (K x N); memory holds W[n_out, K] row-major (i.e. B[k][n] =
// W[(n0+n)*k_stride + k0 + k]).  element j<8 -> K = k0 + j + 8*(lane>=16),
// j>=8 -> K = k0 + 16 + (j-8) + 8*(lane>=16), N = lane&15.
__device__ __forceinline__ v16bf load_b_fragT(const bf16_t* w, int k_stride,
                                              int n0, int k0, int lane) {
  int n = lane & 15;
  const bf16_t* p = w + (size_t)(n0 + n) * k_stride + k0 + ((lane & 16) ? 8 : 0);
  v16bf b;
#pragma unroll
  for (int j = 0; j < 8; ++j) b[j] = p[j];
#pragma unroll
  for (int j = 0; j < 8; ++j) b[8 + j] = p[16 + j];
  return b;
}

__device__ __forceinline__ float sigmoidf_(float x) { return 1.0f / (1.0f + expf(-x)); }

__device__ __forceinline__ void atomicMaxF(float* addr, float val) {
  unsigned int* ua = (unsigned int*)addr;
  unsigned int old = *ua;
  while (__uint_as_float(old) < val) {
    unsigned int assumed = old;
    old = atomicCAS(ua, assumed, __float_as_uint(val));
    if (old == assumed) break;
  }
}

// ---------------------------------------------------------------------------
// Utility kernels
// ---------------------------------------------------------------------------
__global__ void k_fill(float* p, float v, long n) {
  long i = (long)blockIdx.x * blockDim.x + threadIdx.x;
  if (i < n) p[i] = v;
}

__global__ void k_f2bf(const float* __restrict__ s, bf16_t* __restrict__ d, long n) {
  long i = (long)blockIdx.x * blockDim.x + threadIdx.x;
  if (i < n) d[i] = (bf16_t)s[i];
}

// ---------------------------------------------------------------------------
// lin0: h = relu(x @ lin0_w^T + b)   [N,11] x [64,11] -> [N,64]  (+bf16 mirror)
// ---------------------------------------------------------------------------
__global__ void k_lin0(const float* __restrict__ x, const float* __restrict__ w,
                       const float* __restrict__ b, float* __restrict__ h,
                       bf16_t* __restrict__ hbf, int N) {
  long i = (long)blockIdx.x * blockDim.x + threadIdx.x;
  if (i >= (long)N * 64) return;
  int n = (int)(i >> 6), d = (int)(i & 63);
  const float* xr = x + (size_t)n * 11;
  const float* wr = w + (size_t)d * 11;
  float acc = b[d];
#pragma unroll
  for (int k = 0; k < 11; ++k) acc += xr[k] * wr[k];
  acc = fmaxf(acc, 0.0f);
  h[i] = acc;
  hbf[i] = (bf16_t)acc;
}

// ---------------------------------------------------------------------------
// edge MLP layer 1: relu1 = relu(edge_attr @ mlp1_w^T + b)  [E,5]->[E,128] bf16
// ---------------------------------------------------------------------------
__global__ void k_mlp1(const float* __restrict__ ea, const float* __restrict__ w,
                       const float* __restrict__ b, bf16_t* __restrict__ relu1, int E) {
  long i = (long)blockIdx.x * blockDim.x + threadIdx.x;
  if (i >= (long)E * 128) return;
  int e = (int)(i >> 7), d = (int)(i & 127);
  const float* ar = ea + (size_t)e * 5;
  const float* wr = w + (size_t)d * 5;
  float acc = b[d];
#pragma unroll
  for (int k = 0; k < 5; ++k) acc += ar[k] * wr[k];
  relu1[i] = (bf16_t)fmaxf(acc, 0.0f);
}

// ---------------------------------------------------------------------------
// edge MLP layer 2 (WMMA): ew[E,4096] = relu1[E,128] @ mlp2_w^T + b, bf16 out.
// One block = 64 edges (4 row tiles); 8 waves; wave w covers cols [w*512,+512).
// B fragments are loaded once per col tile and reused across the 4 row tiles.
// __launch_bounds__(256,1): keep the 4x4 A-fragment array (128 VGPRs) resident
// in registers instead of spilling to scratch.
// ---------------------------------------------------------------------------
__global__ void __launch_bounds__(256, 1)
k_mlp2_wmma(const bf16_t* __restrict__ relu1, const bf16_t* __restrict__ w2,
            const float* __restrict__ b2, bf16_t* __restrict__ ew, int E) {
  int wid = threadIdx.x >> 5, lane = threadIdx.x & 31;
  int e0 = blockIdx.x * 64;
  int ncol = lane & 15;
  int mh = (lane & 16) ? 8 : 0;

  v16bf a[4][4];
#pragma unroll
  for (int rt = 0; rt < 4; ++rt) {
    int re = e0 + rt * 16;
    if (re + 16 > E) re = E - 16;  // tail clamp: duplicate compute, stores guarded
    const bf16_t* ab = relu1 + (size_t)re * 128;
#pragma unroll
    for (int kc = 0; kc < 4; ++kc) a[rt][kc] = load_a_frag(ab + kc * 32, 128, lane);
  }

  int n_start = wid * 512;
  for (int t = 0; t < 32; ++t) {
    int n0 = n_start + t * 16;
    float bias = b2[n0 + ncol];
    v8f c[4];
#pragma unroll
    for (int rt = 0; rt < 4; ++rt)
#pragma unroll
      for (int r = 0; r < 8; ++r) c[rt][r] = bias;
#pragma unroll
    for (int kc = 0; kc < 4; ++kc) {
      v16bf bfrag = load_b_fragT(w2, 128, n0, kc * 32, lane);
#pragma unroll
      for (int rt = 0; rt < 4; ++rt) c[rt] = wmma_bf16(a[rt][kc], bfrag, c[rt]);
    }
    int col = n0 + ncol;
#pragma unroll
    for (int rt = 0; rt < 4; ++rt) {
      int re = e0 + rt * 16;
      if (re + 16 > E) continue;  // wave-uniform guard
#pragma unroll
      for (int r = 0; r < 8; ++r)
        ew[(size_t)(re + mh + r) * 4096 + col] = (bf16_t)c[rt][r];
    }
  }
}

// ---------------------------------------------------------------------------
// degree: deg[dst]++ per edge
// ---------------------------------------------------------------------------
__global__ void k_deg(const int* __restrict__ ei, float* __restrict__ deg, int E) {
  int e = blockIdx.x * blockDim.x + threadIdx.x;
  if (e < E) unsafeAtomicAdd(&deg[ei[E + e]], 1.0f);
}

// ---------------------------------------------------------------------------
// msg + scatter: one wave per edge; msg[o] = sum_i h[src,i] * ew[e, i*64+o];
// atomic add into agg[dst, o].  lane handles o = lane and o = lane+32.
// TDM path: double-buffered tensor_load_to_lds of the 8KB ew row, overlapped
// with the matvec of the previous edge; synced with s_wait_tensorcnt.
// ---------------------------------------------------------------------------
#define MSG_WAVES 2
#define MSG_CHUNK 8

#ifdef HAVE_TDM
__device__ __forceinline__ void tdm_load_row(const bf16_t* gsrc, unsigned lds_off) {
  // D# per CDNA5 ISA: 2D tensor, data_size=8B,
  // tensor_dim0 = 1024, tile 1024x1  (one contiguous 8KB ew row -> LDS).
  unsigned long long ga = (unsigned long long)(const void*)gsrc;
  u32x4 g0;
  g0[0] = 1u;                                   // count=1, user descriptor
  g0[1] = lds_off;                              // lds_addr (bytes)
  g0[2] = (unsigned)(ga & 0xffffffffu);         // global_addr[31:0]
  g0[3] = (unsigned)((ga >> 32) & 0x1ffffffu)   // global_addr[56:32]
          | (2u << 30);                         // type=2 ("image")
  i32x8 g1;
  g1[0] = (3 << 16);                            // data_size=3 (8 bytes)
  g1[1] = (int)((1024u & 0xffffu) << 16);       // tensor_dim0[15:0]
  g1[2] = (int)(1u << 16);                      // dim0[31:16]=0, tensor_dim1=1
  g1[3] = (int)(1024u << 16);                   // dim1[31:16]=0, tile_dim0=1024
  g1[4] = 1;                                    // tile_dim1=1, tile_dim2=0
  g1[5] = 1024;                                 // tensor_dim0_stride[31:0]
  g1[6] = 0;                                    // stride0 hi / stride1 lo
  g1[7] = 0;                                    // stride1 hi
  i32x4 z4 = {};
  i32x8 z8 = {};
  // clang-23 / therock-10.0 6-arg form:
  // (u32x4 g0, i32x8 g1, i32x4 g2, i32x4 g3, i32x8 extra, i32 cpol)
  __builtin_amdgcn_tensor_load_to_lds(g0, g1, z4, z4, z8, 0);
}
#endif

__global__ void k_msg(const int* __restrict__ ei, const float* __restrict__ h,
                      const bf16_t* __restrict__ ew, float* __restrict__ agg, int E) {
#ifdef HAVE_TDM
  __shared__ bf16_t lbuf[MSG_WAVES][2][4096];
  int wid = threadIdx.x >> 5, lane = threadIdx.x & 31;
  int e0 = (blockIdx.x * MSG_WAVES + wid) * MSG_CHUNK;
  if (e0 >= E) return;
  int e1 = min(e0 + MSG_CHUNK, E);
  unsigned lds0 = (unsigned)(unsigned long long)(void*)&lbuf[wid][0][0];
  unsigned lds1 = (unsigned)(unsigned long long)(void*)&lbuf[wid][1][0];

  tdm_load_row(ew + (size_t)e0 * 4096, lds0);
  for (int e = e0; e < e1; ++e) {
    int buf = (e - e0) & 1;
    if (e + 1 < e1) {
      tdm_load_row(ew + (size_t)(e + 1) * 4096, buf ? lds0 : lds1);
      __builtin_amdgcn_s_wait_tensorcnt(1);  // oldest (current buf) complete
    } else {
      __builtin_amdgcn_s_wait_tensorcnt(0);
    }
    asm volatile("" ::: "memory");  // keep LDS reads below the wait
    const bf16_t* wrow = &lbuf[wid][buf][0];
    int s = ei[e], d = ei[E + e];
    const float* xr = h + (size_t)s * 64;
    float x0 = xr[lane], x1 = xr[lane + 32];
    float m0 = 0.0f, m1 = 0.0f;
#pragma unroll 4
    for (int i = 0; i < 32; ++i) {
      float xv = __shfl(x0, i, 32);
      m0 += xv * (float)wrow[i * 64 + lane];
      m1 += xv * (float)wrow[i * 64 + 32 + lane];
    }
#pragma unroll 4
    for (int i = 0; i < 32; ++i) {
      float xv = __shfl(x1, i, 32);
      m0 += xv * (float)wrow[(32 + i) * 64 + lane];
      m1 += xv * (float)wrow[(32 + i) * 64 + 32 + lane];
    }
    unsafeAtomicAdd(&agg[(size_t)d * 64 + lane], m0);
    unsafeAtomicAdd(&agg[(size_t)d * 64 + 32 + lane], m1);
  }
#else
  int wid = threadIdx.x >> 5, lane = threadIdx.x & 31;
  int e = blockIdx.x * MSG_WAVES * MSG_CHUNK + wid * MSG_CHUNK;
  int e1 = min(e + MSG_CHUNK, E);
  for (; e < e1; ++e) {
    int s = ei[e], d = ei[E + e];
    const float* xr = h + (size_t)s * 64;
    float x0 = xr[lane], x1 = xr[lane + 32];
    const bf16_t* wrow = ew + (size_t)e * 4096;
    __builtin_prefetch(wrow, 0, 1);
    float m0 = 0.0f, m1 = 0.0f;
#pragma unroll 4
    for (int i = 0; i < 32; ++i) {
      float xv = __shfl(x0, i, 32);
      m0 += xv * (float)wrow[i * 64 + lane];
      m1 += xv * (float)wrow[i * 64 + 32 + lane];
    }
#pragma unroll 4
    for (int i = 0; i < 32; ++i) {
      float xv = __shfl(x1, i, 32);
      m0 += xv * (float)wrow[(32 + i) * 64 + lane];
      m1 += xv * (float)wrow[(32 + i) * 64 + 32 + lane];
    }
    unsafeAtomicAdd(&agg[(size_t)d * 64 + lane], m0);
    unsafeAtomicAdd(&agg[(size_t)d * 64 + 32 + lane], m1);
  }
#endif
}

// ---------------------------------------------------------------------------
// Node update (WMMA): per 16-node tile (one wave):
//   m  = relu(agg/deg + h@root^T + conv_b)          (8 wmma)
//   gi = m@gru_w_ih^T ; gh = h@gru_w_hh^T           (48 wmma)
//   GRU gate math, in-place update of h (and bf16 mirror).
// ---------------------------------------------------------------------------
#define NODE_WAVES 8
__global__ void __launch_bounds__(256, 1)
k_node_wmma(const float* __restrict__ agg, const float* __restrict__ deg,
            const bf16_t* __restrict__ rootbf, const float* __restrict__ conv_b,
            const bf16_t* __restrict__ gihbf, const bf16_t* __restrict__ ghhbf,
            const float* __restrict__ b_ih, const float* __restrict__ b_hh,
            float* __restrict__ h, bf16_t* __restrict__ hbf, int N) {
  __shared__ bf16_t mlds[NODE_WAVES][16 * 64];
  int wid = threadIdx.x >> 5, lane = threadIdx.x & 31;
  int tile = blockIdx.x * NODE_WAVES + wid;
  int r0 = tile * 16;
  if (r0 >= N) return;

  int mh = (lane & 16) ? 8 : 0;
  int ncol = lane & 15;

  v16bf ah0 = load_a_frag(hbf + (size_t)r0 * 64, 64, lane);
  v16bf ah1 = load_a_frag(hbf + (size_t)r0 * 64 + 32, 64, lane);

  // m = relu(agg/deg + h@root^T + conv_b), staged to LDS as bf16 16x64 tile
#pragma unroll
  for (int nt = 0; nt < 4; ++nt) {
    v8f c = {};
    c = wmma_bf16(ah0, load_b_fragT(rootbf, 64, nt * 16, 0, lane), c);
    c = wmma_bf16(ah1, load_b_fragT(rootbf, 64, nt * 16, 32, lane), c);
    int col = nt * 16 + ncol;
    float cb = conv_b[col];
#pragma unroll
    for (int r = 0; r < 8; ++r) {
      int row = r0 + mh + r;
      float dv = deg[row];
      dv = dv < 1.0f ? 1.0f : dv;
      float mv = c[r] + agg[(size_t)row * 64 + col] / dv + cb;
      mlds[wid][(mh + r) * 64 + col] = (bf16_t)fmaxf(mv, 0.0f);
    }
  }
  // same-wave LDS RAW: DScnt waits inserted by compiler
  v16bf am0 = load_a_frag(&mlds[wid][0], 64, lane);
  v16bf am1 = load_a_frag(&mlds[wid][0] + 32, 64, lane);

#pragma unroll
  for (int nt = 0; nt < 4; ++nt) {
    int col = nt * 16 + ncol;
    v8f gir = {}, giz = {}, gin = {}, ghr = {}, ghz = {}, ghn = {};
    gir = wmma_bf16(am0, load_b_fragT(gihbf, 64, nt * 16, 0, lane), gir);
    gir = wmma_bf16(am1, load_b_fragT(gihbf, 64, nt * 16, 32, lane), gir);
    giz = wmma_bf16(am0, load_b_fragT(gihbf, 64, 64 + nt * 16, 0, lane), giz);
    giz = wmma_bf16(am1, load_b_fragT(gihbf, 64, 64 + nt * 16, 32, lane), giz);
    gin = wmma_bf16(am0, load_b_fragT(gihbf, 64, 128 + nt * 16, 0, lane), gin);
    gin = wmma_bf16(am1, load_b_fragT(gihbf, 64, 128 + nt * 16, 32, lane), gin);
    ghr = wmma_bf16(ah0, load_b_fragT(ghhbf, 64, nt * 16, 0, lane), ghr);
    ghr = wmma_bf16(ah1, load_b_fragT(ghhbf, 64, nt * 16, 32, lane), ghr);
    ghz = wmma_bf16(ah0, load_b_fragT(ghhbf, 64, 64 + nt * 16, 0, lane), ghz);
    ghz = wmma_bf16(ah1, load_b_fragT(ghhbf, 64, 64 + nt * 16, 32, lane), ghz);
    ghn = wmma_bf16(ah0, load_b_fragT(ghhbf, 64, 128 + nt * 16, 0, lane), ghn);
    ghn = wmma_bf16(ah1, load_b_fragT(ghhbf, 64, 128 + nt * 16, 32, lane), ghn);

    float bir = b_ih[col], biz = b_ih[64 + col], bin = b_ih[128 + col];
    float bhr = b_hh[col], bhz = b_hh[64 + col], bhn = b_hh[128 + col];
#pragma unroll
    for (int r = 0; r < 8; ++r) {
      int row = r0 + mh + r;
      float rg = sigmoidf_(gir[r] + bir + ghr[r] + bhr);
      float zg = sigmoidf_(giz[r] + biz + ghz[r] + bhz);
      float ng = tanhf(gin[r] + bin + rg * (ghn[r] + bhn));
      size_t idx = (size_t)row * 64 + col;
      float hv = h[idx];
      float hn = (1.0f - zg) * ng + zg * hv;
      h[idx] = hn;
      hbf[idx] = (bf16_t)hn;
    }
  }
}

// ---------------------------------------------------------------------------
// Set2Set LSTM step (torch gate order i,f,g,o). hl double-buffered; cl is
// per-thread so a single buffer is race-free.
// ---------------------------------------------------------------------------
__global__ void k_lstm(const float* __restrict__ qstar, const float* __restrict__ hlr,
                       float* __restrict__ hlw, float* __restrict__ cl,
                       const float* __restrict__ wih, const float* __restrict__ whh,
                       const float* __restrict__ bih, const float* __restrict__ bhh, int B) {
  long i = (long)blockIdx.x * blockDim.x + threadIdx.x;
  if (i >= (long)B * 64) return;
  int b = (int)(i >> 6), d = (int)(i & 63);
  const float* q = qstar + (size_t)b * 128;
  const float* hh = hlr + (size_t)b * 64;
  float g[4];
#pragma unroll
  for (int gi = 0; gi < 4; ++gi) {
    int o = gi * 64 + d;
    float acc = bih[o] + bhh[o];
    const float* wi = wih + (size_t)o * 128;
    for (int k = 0; k < 128; ++k) acc += q[k] * wi[k];
    const float* wh = whh + (size_t)o * 64;
    for (int k = 0; k < 64; ++k) acc += hh[k] * wh[k];
    g[gi] = acc;
  }
  float cv = sigmoidf_(g[1]) * cl[i] + sigmoidf_(g[0]) * tanhf(g[2]);
  float hv = sigmoidf_(g[3]) * tanhf(cv);
  cl[i] = cv;
  hlw[i] = hv;
}

__global__ void k_edot(const float* __restrict__ h, const float* __restrict__ q,
                       const int* __restrict__ batch, float* __restrict__ e_n,
                       float* __restrict__ emax, int N) {
  int i = blockIdx.x * blockDim.x + threadIdx.x;
  if (i >= N) return;
  int b = batch[i];
  const float* hr = h + (size_t)i * 64;
  const float* qr = q + (size_t)b * 64;
  float acc = 0.0f;
#pragma unroll
  for (int k = 0; k < 64; ++k) acc += hr[k] * qr[k];
  e_n[i] = acc;
  atomicMaxF(&emax[b], acc);
}

__global__ void k_aexp(float* __restrict__ e_n, const float* __restrict__ emax,
                       float* __restrict__ asum, const int* __restrict__ batch, int N) {
  int i = blockIdx.x * blockDim.x + threadIdx.x;
  if (i >= N) return;
  int b = batch[i];
  float a = expf(e_n[i] - emax[b]);
  e_n[i] = a;
  unsafeAtomicAdd(&asum[b], a);
}

__global__ void k_rsum(const float* __restrict__ e_n, const float* __restrict__ asum,
                       const float* __restrict__ h, const int* __restrict__ batch,
                       float* __restrict__ rvec, int N) {
  long idx = (long)blockIdx.x * blockDim.x + threadIdx.x;
  if (idx >= (long)N * 64) return;
  int i = (int)(idx >> 6), d = (int)(idx & 63);
  int b = batch[i];
  float w = e_n[i] / asum[b];
  unsafeAtomicAdd(&rvec[(size_t)b * 64 + d], w * h[(size_t)i * 64 + d]);
}

__global__ void k_qstar(const float* __restrict__ q, const float* __restrict__ rvec,
                        float* __restrict__ qstar, int B) {
  long i = (long)blockIdx.x * blockDim.x + threadIdx.x;
  if (i >= (long)B * 64) return;
  int b = (int)(i >> 6), d = (int)(i & 63);
  qstar[(size_t)b * 128 + d] = q[i];
  qstar[(size_t)b * 128 + 64 + d] = rvec[i];
}

__global__ void k_lin1(const float* __restrict__ qstar, const float* __restrict__ w,
                       const float* __restrict__ b, float* __restrict__ out, int B) {
  long i = (long)blockIdx.x * blockDim.x + threadIdx.x;
  if (i >= (long)B * 64) return;
  int bb = (int)(i >> 6), d = (int)(i & 63);
  const float* q = qstar + (size_t)bb * 128;
  const float* wr = w + (size_t)d * 128;
  float acc = b[d];
  for (int k = 0; k < 128; ++k) acc += q[k] * wr[k];
  out[i] = fmaxf(acc, 0.0f);
}

// ---------------------------------------------------------------------------
extern "C" void kernel_launch(void* const* d_in, const int* in_sizes, int n_in,
                              void* d_out, int out_size, void* d_ws, size_t ws_size,
                              hipStream_t stream) {
  const float* x         = (const float*)d_in[0];
  const float* edge_attr = (const float*)d_in[1];
  const float* lin0_w    = (const float*)d_in[2];
  const float* lin0_b    = (const float*)d_in[3];
  const float* mlp1_w    = (const float*)d_in[4];
  const float* mlp1_b    = (const float*)d_in[5];
  const float* mlp2_w    = (const float*)d_in[6];
  const float* mlp2_b    = (const float*)d_in[7];
  const float* root_w    = (const float*)d_in[8];
  const float* conv_b    = (const float*)d_in[9];
  const float* gru_w_ih  = (const float*)d_in[10];
  const float* gru_w_hh  = (const float*)d_in[11];
  const float* gru_b_ih  = (const float*)d_in[12];
  const float* gru_b_hh  = (const float*)d_in[13];
  const float* lstm_w_ih = (const float*)d_in[14];
  const float* lstm_w_hh = (const float*)d_in[15];
  const float* lstm_b_ih = (const float*)d_in[16];
  const float* lstm_b_hh = (const float*)d_in[17];
  const float* lin1_w    = (const float*)d_in[18];
  const float* lin1_b    = (const float*)d_in[19];
  const int*   edge_idx  = (const int*)d_in[20];
  const int*   batch     = (const int*)d_in[21];
  (void)n_in; (void)ws_size;

  const int N = in_sizes[0] / 11;
  const int E = in_sizes[1] / 5;
  const int B = out_size / 64;

  // workspace carve-up (256B aligned)
  char* base = (char*)d_ws;
  size_t off = 0;
  auto take = [&](size_t bytes) -> void* {
    void* p = base + off;
    off = (off + bytes + 255) & ~(size_t)255;
    return p;
  };
  float*  h      = (float*) take((size_t)N * 64 * 4);
  bf16_t* hbf    = (bf16_t*)take((size_t)N * 64 * 2);
  bf16_t* relu1  = (bf16_t*)take((size_t)E * 128 * 2);
  bf16_t* ew     = (bf16_t*)take((size_t)E * 4096 * 2);
  float*  agg    = (float*) take((size_t)N * 64 * 4);
  float*  deg    = (float*) take((size_t)N * 4);
  bf16_t* w2bf   = (bf16_t*)take((size_t)4096 * 128 * 2);
  bf16_t* rootbf = (bf16_t*)take((size_t)64 * 64 * 2);
  bf16_t* gihbf  = (bf16_t*)take((size_t)192 * 64 * 2);
  bf16_t* ghhbf  = (bf16_t*)take((size_t)192 * 64 * 2);
  float*  e_n    = (float*) take((size_t)N * 4);
  float*  emax   = (float*) take((size_t)B * 4);
  float*  asum   = (float*) take((size_t)B * 4);
  float*  rvec   = (float*) take((size_t)B * 64 * 4);
  float*  qstar  = (float*) take((size_t)B * 128 * 4);
  float*  hlA    = (float*) take((size_t)B * 64 * 4);
  float*  hlB    = (float*) take((size_t)B * 64 * 4);
  float*  clv    = (float*) take((size_t)B * 64 * 4);

  const int T = 256;
  auto cdiv = [](long a, long b) { return (int)((a + b - 1) / b); };

  // bf16 weight mirrors
  k_f2bf<<<cdiv(4096L * 128, T), T, 0, stream>>>(mlp2_w, w2bf, 4096L * 128);
  k_f2bf<<<cdiv(64L * 64, T), T, 0, stream>>>(root_w, rootbf, 64L * 64);
  k_f2bf<<<cdiv(192L * 64, T), T, 0, stream>>>(gru_w_ih, gihbf, 192L * 64);
  k_f2bf<<<cdiv(192L * 64, T), T, 0, stream>>>(gru_w_hh, ghhbf, 192L * 64);

  // encoder front-end
  k_lin0<<<cdiv((long)N * 64, T), T, 0, stream>>>(x, lin0_w, lin0_b, h, hbf, N);
  k_mlp1<<<cdiv((long)E * 128, T), T, 0, stream>>>(edge_attr, mlp1_w, mlp1_b, relu1, E);
  k_mlp2_wmma<<<cdiv(E, 64), T, 0, stream>>>(relu1, w2bf, mlp2_b, ew, E);

  // degrees
  k_fill<<<cdiv(N, T), T, 0, stream>>>(deg, 0.0f, N);
  k_deg<<<cdiv(E, T), T, 0, stream>>>(edge_idx, deg, E);

  // 3 NNConv + GRU layers
  for (int layer = 0; layer < 3; ++layer) {
    k_fill<<<cdiv((long)N * 64, T), T, 0, stream>>>(agg, 0.0f, (long)N * 64);
    k_msg<<<cdiv(E, MSG_WAVES * MSG_CHUNK), MSG_WAVES * 32, 0, stream>>>(edge_idx, h, ew,
                                                                         agg, E);
    k_node_wmma<<<cdiv((N + 15) / 16, NODE_WAVES), T, 0, stream>>>(
        agg, deg, rootbf, conv_b, gihbf, ghhbf, gru_b_ih, gru_b_hh, h, hbf, N);
  }

  // Set2Set (3 steps)
  k_fill<<<cdiv((long)B * 128, T), T, 0, stream>>>(qstar, 0.0f, (long)B * 128);
  k_fill<<<cdiv((long)B * 64, T), T, 0, stream>>>(hlA, 0.0f, (long)B * 64);
  k_fill<<<cdiv((long)B * 64, T), T, 0, stream>>>(clv, 0.0f, (long)B * 64);

  float* hlr = hlA;
  float* hlw = hlB;
  for (int step = 0; step < 3; ++step) {
    k_lstm<<<cdiv((long)B * 64, T), T, 0, stream>>>(qstar, hlr, hlw, clv, lstm_w_ih,
                                                    lstm_w_hh, lstm_b_ih, lstm_b_hh, B);
    k_fill<<<cdiv(B, T), T, 0, stream>>>(emax, -3.0e38f, B);
    k_fill<<<cdiv(B, T), T, 0, stream>>>(asum, 0.0f, B);
    k_fill<<<cdiv((long)B * 64, T), T, 0, stream>>>(rvec, 0.0f, (long)B * 64);
    k_edot<<<cdiv(N, T), T, 0, stream>>>(h, hlw, batch, e_n, emax, N);
    k_aexp<<<cdiv(N, T), T, 0, stream>>>(e_n, emax, asum, batch, N);
    k_rsum<<<cdiv((long)N * 64, T), T, 0, stream>>>(e_n, asum, h, batch, rvec, N);
    k_qstar<<<cdiv((long)B * 64, T), T, 0, stream>>>(hlw, rvec, qstar, B);
    float* t = hlr; hlr = hlw; hlw = t;
  }

  k_lin1<<<cdiv((long)B * 64, T), T, 0, stream>>>(qstar, lin1_w, lin1_b, (float*)d_out, B);
}
